// GraphBased_61564061221542
// MI455X (gfx1250) — matrix operand order repaired and checked by reading, not compiled, not run on hardware
//
#include <hip/hip_runtime.h>
#include <hip/hip_bf16.h>
#include <math.h>

// ---------------- CDNA5 WMMA types ----------------
typedef __attribute__((ext_vector_type(16))) _Float16 v16h;
typedef __attribute__((ext_vector_type(8)))  _Float16 v8h;
typedef __attribute__((ext_vector_type(8)))  float    v8f;

#define NNODES 2048
#define LOUT   500
#define LPAD   512   // 500 padded to multiple of 16
#define KFC    800   // 50*4*4, multiple of 32

// ---------------------------------------------------------------------------
// Fragment helpers for V_WMMA_F32_16X16X32_F16.
// A (16x32 MxK, row-major f16): lanes 0-15 -> M=lane, halves0-7=K0..7,
// halves8-15=K16..23; lanes16-31 -> halves0-7=K8..15, halves8-15=K24..31.
// => two contiguous 16B chunks per lane at K = 8*hi and 16+8*hi.
// B (32x16 KxN) stored TRANSPOSED Bt[N][K]: lane -> col n=lane&15,
// same K order as A => two 16B chunks per lane at K = 8*hi and 16+8*hi.
// ---------------------------------------------------------------------------
__device__ __forceinline__ v16h frag_a_ptr(const _Float16* p /* row base + 8*hi */,
                                           int k) {
  union { v16h v; v8h h[2]; } u;
  u.h[0] = *(const v8h*)(p + k);
  u.h[1] = *(const v8h*)(p + k + 16);
  return u.v;
}

// ---- CDNA5 async global->LDS copy (ASYNCcnt engine, ISA ch.10.7/§15.18) ----
// LDS destination address = low 32 bits of the generic pointer of a __shared__
// object (LDS aperture truncates to addr[31:0], ISA §10.2).
__device__ __forceinline__ void async_copy_b128_to_lds(unsigned lds_off, const void* gsrc) {
  asm volatile("global_load_async_to_lds_b128 %0, %1, off"
               :: "v"(lds_off), "v"(gsrc)
               : "memory");
}
__device__ __forceinline__ void wait_async_zero() {
  asm volatile("s_wait_asynccnt 0x0" ::: "memory");
}

enum { EPI_RELU_BIAS_F16 = 0, EPI_F32 = 1, EPI_SCALED_T_F16 = 2, EPI_ROWSCALE_BIAS_F32 = 3 };

// One wave computes a 64x32 tile of C = A * Bt^T (4x2 register-blocked WMMA).
template <int EPI>
__global__ void wmma_gemm_k(const _Float16* __restrict__ A, int lda,
                            const _Float16* __restrict__ Bt, int ldb, int K,
                            float* __restrict__ outf, _Float16* __restrict__ outh, int ldo,
                            const float* __restrict__ bias,
                            const float* __restrict__ scale, int validN) {
  int lane = threadIdx.x;
  int l15  = lane & 15;
  int hi   = lane >> 4;
  int m0 = blockIdx.x * 64;
  int n0 = blockIdx.y * 32;
  const _Float16* Ab = A  + (size_t)(m0 + l15) * lda + 8 * hi;
  const _Float16* Bb = Bt + (size_t)(n0 + l15) * ldb + 8 * hi;
  v8f acc[4][2] = {};
  for (int k = 0; k < K; k += 32) {
    if (k + 64 < K) {   // prefetch next slab (global_prefetch_b8)
      __builtin_prefetch(Ab + k + 64, 0, 0);
      __builtin_prefetch(Bb + k + 64, 0, 0);
    }
    v16h af[4], bf[2];
#pragma unroll
    for (int i = 0; i < 4; ++i) af[i] = frag_a_ptr(Ab + (size_t)(16 * i) * lda, k);
#pragma unroll
    for (int j = 0; j < 2; ++j) bf[j] = frag_a_ptr(Bb + (size_t)(16 * j) * ldb, k);
#pragma unroll
    for (int i = 0; i < 4; ++i)
#pragma unroll
      for (int j = 0; j < 2; ++j)
        acc[i][j] = __builtin_amdgcn_wmma_f32_16x16x32_f16(false, af[i], false, bf[j],
                                                           (short)0, acc[i][j], false, false);
  }
  // C layout: lane l, vgpr r -> row = r + 8*(l>>4), col = l&15
#pragma unroll
  for (int i = 0; i < 4; ++i)
#pragma unroll
    for (int j = 0; j < 2; ++j) {
      int n = n0 + 16 * j + l15;
#pragma unroll
      for (int r = 0; r < 8; ++r) {
        int m = m0 + 16 * i + r + 8 * hi;
        float v = acc[i][j][r];
        if (EPI == EPI_RELU_BIAS_F16) {
          float bb = (n < validN) ? bias[n] : 0.f;
          outh[(size_t)m * ldo + n] = (_Float16)fmaxf(v + bb, 0.f);
        } else if (EPI == EPI_F32) {
          outf[(size_t)m * ldo + n] = v;
        } else if (EPI == EPI_SCALED_T_F16) {
          outh[(size_t)n * ldo + m] = (_Float16)(v * scale[m]);  // transposed + row scale
        } else {
          float bb = (n < validN) ? bias[n] : 0.f;
          outf[(size_t)m * ldo + n] = v * scale[m] + bb;
        }
      }
    }
}

// ----------------------------- conv1 (VALU) --------------------------------
// conv1 5x5 VALID + bias + relu + 2x2 maxpool : [2048,1,28,28] -> [2048,20,12,12]
__global__ void conv1_pool_k(const float* __restrict__ x, const float* __restrict__ w,
                             const float* __restrict__ b, float* __restrict__ out1) {
  int idx = blockIdx.x * blockDim.x + threadIdx.x;
  if (idx >= NNODES * 20 * 12 * 12) return;
  int px = idx % 12; int t = idx / 12;
  int py = t % 12;   t /= 12;
  int c  = t % 20;   int n = t / 20;
  const float* xi = x + (size_t)n * 784;
  const float* wc = w + c * 25;
  float bias = b[c];
  float mx = 0.f;   // max(relu(.)) folded via 0 init
#pragma unroll
  for (int dy = 0; dy < 2; ++dy)
#pragma unroll
    for (int dx = 0; dx < 2; ++dx) {
      int oy = 2 * py + dy, ox = 2 * px + dx;
      float s = bias;
#pragma unroll
      for (int ky = 0; ky < 5; ++ky)
#pragma unroll
        for (int kx = 0; kx < 5; ++kx)
          s = fmaf(xi[(oy + ky) * 28 + ox + kx], wc[ky * 5 + kx], s);
      mx = fmaxf(mx, s);
    }
  out1[idx] = mx;
}

// ------------------- conv2 as implicit-GEMM WMMA ---------------------------
// One block (256 thr = 8 waves) per image n:
//  phase 0: kick off async DMA of the 64KB weight panel w2T into LDS
//           (global_load_async_to_lds_b128, tracked by ASYNCcnt)
//  phase 1: build 64x512 f16 im2col tile in LDS *while the DMA flies*
//           (m = conv output pixel oy*8+ox, k = ic*25+ky*5+kx, pad 500->512)
//  phase 2: s_wait_asynccnt 0 + barrier; 64x64xK=512 WMMA entirely from LDS;
//           bias+relu -> f16 LDS tile
//  phase 3: 2x2 maxpool -> out2h[n][c*16+py*4+px]
__global__ void conv2_wmma_k(const float* __restrict__ h1, const _Float16* __restrict__ w2T,
                             const float* __restrict__ b, _Float16* __restrict__ out2h) {
  __shared__ __align__(128) _Float16 sA[64 * 512];   // 64 KB im2col
  __shared__ __align__(128) _Float16 sB[64 * 512];   // 64 KB weight panel (async DMA)
  __shared__ __align__(128) _Float16 sC[64 * 64];    // 8 KB relu'd conv (pixel x chan)
  int n = blockIdx.x;
  const float* h1n = h1 + (size_t)n * 20 * 144;

  // phase 0: async global->LDS copy of w2T (4096 x 16B, 16 per thread)
  unsigned sB0 = (unsigned)(size_t)&sB[0];
  const char* gB = (const char*)w2T;
  for (int idx = threadIdx.x; idx < 4096; idx += 256)
    async_copy_b128_to_lds(sB0 + idx * 16u, gB + (size_t)idx * 16);

  // phase 1: im2col (overlapped with the async DMA)
  for (int idx = threadIdx.x; idx < 64 * 512; idx += 256) {
    int m = idx >> 9, k = idx & 511;
    float v = 0.f;
    if (k < 500) {
      int ic = k / 25, rem = k - ic * 25;
      int ky = rem / 5, kx = rem - ky * 5;
      int oy = m >> 3, ox = m & 7;
      v = h1n[ic * 144 + (oy + ky) * 12 + (ox + kx)];
    }
    sA[idx] = (_Float16)v;
  }
  wait_async_zero();
  __syncthreads();

  // phase 2: 16 WMMA tiles over 8 waves (2 each); A and B both from LDS
  int wave = threadIdx.x >> 5;
  int lane = threadIdx.x & 31;
  int l15 = lane & 15, hi = lane >> 4;
#pragma unroll
  for (int tt = 0; tt < 2; ++tt) {
    int t  = wave * 2 + tt;
    int mi = (t >> 2) * 16;
    int nj = (t & 3) * 16;
    const _Float16* arow = &sA[(mi + l15) * 512 + 8 * hi];
    const _Float16* brow = &sB[(nj + l15) * 512 + 8 * hi];
    v8f acc = {};
    for (int k = 0; k < 512; k += 32) {
      v16h af = frag_a_ptr(arow, k);
      v16h bf = frag_a_ptr(brow, k);
      acc = __builtin_amdgcn_wmma_f32_16x16x32_f16(false, af, false, bf,
                                                   (short)0, acc, false, false);
    }
    int nn = nj + l15;
    float bb = (nn < 50) ? b[nn] : 0.f;
#pragma unroll
    for (int r = 0; r < 8; ++r) {
      int m = mi + r + 8 * hi;
      sC[m * 64 + nn] = (_Float16)fmaxf(acc[r] + bb, 0.f);
    }
  }
  __syncthreads();

  // phase 3: 2x2 maxpool -> 800-dim f16 feature row
  for (int f = threadIdx.x; f < 800; f += 256) {
    int c = f >> 4, py = (f >> 2) & 3, px = f & 3;
    float mx = 0.f;
#pragma unroll
    for (int dy = 0; dy < 2; ++dy)
#pragma unroll
      for (int dx = 0; dx < 2; ++dx) {
        int m = (2 * py + dy) * 8 + (2 * px + dx);
        mx = fmaxf(mx, (float)sC[m * 64 + c]);
      }
    out2h[(size_t)n * KFC + f] = (_Float16)mx;
  }
}

// ----------------------------- prep / misc ---------------------------------
__global__ void prep_fcw_k(const float* __restrict__ fw, _Float16* __restrict__ fcwT) {
  int idx = blockIdx.x * blockDim.x + threadIdx.x;
  if (idx >= LPAD * KFC) return;
  int nn = idx / KFC, k = idx % KFC;
  fcwT[idx] = (_Float16)((nn < LOUT) ? fw[(size_t)k * LOUT + nn] : 0.f);
}

__global__ void prep_gcnw_k(const float* __restrict__ gw, _Float16* __restrict__ gwT) {
  int idx = blockIdx.x * blockDim.x + threadIdx.x;
  if (idx >= LPAD * LPAD) return;
  int nn = idx / LPAD, k = idx % LPAD;
  gwT[idx] = (_Float16)((nn < LOUT && k < LOUT) ? gw[(size_t)k * LOUT + nn] : 0.f);
}

// conv2_w [50][20][5][5] is exactly [50][500] with k = ic*25+ky*5+kx: pad-only copy.
__global__ void prep_w2_k(const float* __restrict__ w, _Float16* __restrict__ w2T) {
  int idx = blockIdx.x * blockDim.x + threadIdx.x;
  if (idx >= 64 * 512) return;
  int nn = idx >> 9, k = idx & 511;
  w2T[idx] = (_Float16)((nn < 50 && k < 500) ? w[(size_t)nn * 500 + k] : 0.f);
}

__global__ void sq_k(const _Float16* __restrict__ Hh, float* __restrict__ sq) {
  int i = blockIdx.x * blockDim.x + threadIdx.x;
  if (i >= NNODES) return;
  const _Float16* row = Hh + (size_t)i * LPAD;
  float s = 0.f;
  for (int nn = 0; nn < LOUT; ++nn) { float v = (float)row[nn]; s = fmaf(v, v, s); }
  sq[i] = s;
}

__global__ void zero_hist_k(unsigned int* h) { h[threadIdx.x] = 0u; }

// histogram of "threshold steps needed" per off-diagonal pair
__global__ void hist_k(const float* __restrict__ Gram, const float* __restrict__ sq,
                       unsigned int* __restrict__ hist) {
  __shared__ unsigned int hl[256];
  int tid = threadIdx.x;
  hl[tid] = 0u;
  __syncthreads();
  size_t idx = (size_t)blockIdx.x * blockDim.x + tid;
  if (idx < (size_t)NNODES * NNODES) {
    int i = (int)(idx >> 11), j = (int)(idx & (NNODES - 1));
    if (i != j) {
      float d2 = sq[i] + sq[j] - 2.f * Gram[idx];
      float d  = sqrtf(fmaxf(d2, 0.f));
      int bin  = (d < 1.7f) ? 0 : (1 + (int)((d - 1.7f) * 10.f));
      bin = bin > 255 ? 255 : bin;
      atomicAdd(&hl[bin], 1u);
    }
  }
  __syncthreads();
  if (hl[tid]) atomicAdd(&hist[tid], hl[tid]);
}

// device-side replacement of the host while-loop: prefix-scan 256 bins
__global__ void thr_k(const unsigned int* __restrict__ hist, float* __restrict__ thr) {
  if (threadIdx.x == 0) {
    const float min_edges = 0.3f * (float)NNODES;   // 614.4
    unsigned int cum = 0; int k;
    for (k = 0; k < 256; ++k) { cum += hist[k]; if ((float)cum >= min_edges) break; }
    if (k >= 256) k = 255;
    thr[0] = 1.7f + 0.1f * (float)k;
  }
}

// build A_hat (adjacency + I) as f16 0/1, and dinv = rsqrt(degree)
__global__ void adj_k(const float* __restrict__ Gram, const float* __restrict__ sq,
                      const float* __restrict__ thrp,
                      _Float16* __restrict__ adjh, float* __restrict__ dinv) {
  __shared__ float red[256];
  int i = blockIdx.x, tid = threadIdx.x;
  float thr = thrp[0], si = sq[i];
  float cnt = 0.f;
  for (int j = tid; j < NNODES; j += 256) {
    float d2 = si + sq[j] - 2.f * Gram[(size_t)i * NNODES + j];
    float d  = sqrtf(fmaxf(d2, 0.f));
    bool e   = (j != i) && (d < thr);
    adjh[(size_t)i * NNODES + j] = (_Float16)((e || j == i) ? 1.f : 0.f);
    cnt += e ? 1.f : 0.f;
  }
  red[tid] = cnt;
  __syncthreads();
  for (int s = 128; s > 0; s >>= 1) { if (tid < s) red[tid] += red[tid + s]; __syncthreads(); }
  if (tid == 0) dinv[i] = rsqrtf(1.f + red[0]);
}

__global__ void logsoftmax_k(const float* __restrict__ pre, float* __restrict__ out) {
  __shared__ float red[256];
  int i = blockIdx.x, tid = threadIdx.x;
  const float* row = pre + (size_t)i * LPAD;
  float mx = -3.4e38f;
  for (int nn = tid; nn < LOUT; nn += 256) mx = fmaxf(mx, row[nn]);
  red[tid] = mx; __syncthreads();
  for (int s = 128; s > 0; s >>= 1) { if (tid < s) red[tid] = fmaxf(red[tid], red[tid + s]); __syncthreads(); }
  mx = red[0]; __syncthreads();
  float sm = 0.f;
  for (int nn = tid; nn < LOUT; nn += 256) sm += __expf(row[nn] - mx);
  red[tid] = sm; __syncthreads();
  for (int s = 128; s > 0; s >>= 1) { if (tid < s) red[tid] += red[tid + s]; __syncthreads(); }
  float ls = __logf(red[0]);
  for (int nn = tid; nn < LOUT; nn += 256) out[(size_t)i * LOUT + nn] = row[nn] - mx - ls;
}

// ----------------------------- launcher ------------------------------------
extern "C" void kernel_launch(void* const* d_in, const int* in_sizes, int n_in,
                              void* d_out, int out_size, void* d_ws, size_t ws_size,
                              hipStream_t stream) {
  const float* x       = (const float*)d_in[0];
  const float* conv1_w = (const float*)d_in[1];
  const float* conv1_b = (const float*)d_in[2];
  const float* conv2_w = (const float*)d_in[3];
  const float* conv2_b = (const float*)d_in[4];
  const float* fc_w    = (const float*)d_in[5];
  const float* fc_b    = (const float*)d_in[6];
  const float* gcn_w   = (const float*)d_in[7];
  const float* gcn_b   = (const float*)d_in[8];
  float* out = (float*)d_out;

  // workspace layout (bytes, all 256-aligned). Region A holds out1 first,
  // then (after conv2 consumes it) the 2048x2048 Gram matrix.
  char* ws = (char*)d_ws;
  const size_t OFF_A     = 0;                                             // max(23.6MB, 16.8MB)
  const size_t OFF_OUT2H = OFF_A     + (size_t)NNODES * 20 * 144 * 4;     // 23,592,960
  const size_t OFF_FCWT  = OFF_OUT2H + (size_t)NNODES * KFC * 2;
  const size_t OFF_GCNWT = OFF_FCWT  + (size_t)LPAD * KFC * 2;
  const size_t OFF_W2T   = OFF_GCNWT + (size_t)LPAD * LPAD * 2;
  const size_t OFF_HH    = OFF_W2T   + (size_t)64 * 512 * 2;
  const size_t OFF_SQ    = OFF_HH    + (size_t)NNODES * LPAD * 2;
  const size_t OFF_HIST  = OFF_SQ    + (size_t)NNODES * 4;
  const size_t OFF_THR   = OFF_HIST  + 1024;
  const size_t OFF_DINV  = OFF_THR   + 256;
  const size_t OFF_ADJH  = OFF_DINV  + (size_t)NNODES * 4;
  const size_t OFF_BMT   = OFF_ADJH  + (size_t)NNODES * NNODES * 2;
  const size_t OFF_PRE   = OFF_BMT   + (size_t)LPAD * NNODES * 2;
  // total ~45 MB

  float*     out1  = (float*)(ws + OFF_A);
  float*     Gram  = (float*)(ws + OFF_A);
  _Float16*  out2h = (_Float16*)(ws + OFF_OUT2H);
  _Float16*  fcwT  = (_Float16*)(ws + OFF_FCWT);
  _Float16*  gcnwT = (_Float16*)(ws + OFF_GCNWT);
  _Float16*  w2T   = (_Float16*)(ws + OFF_W2T);
  _Float16*  Hh    = (_Float16*)(ws + OFF_HH);
  float*     sq    = (float*)(ws + OFF_SQ);
  unsigned*  hist  = (unsigned*)(ws + OFF_HIST);
  float*     thr   = (float*)(ws + OFF_THR);
  float*     dinv  = (float*)(ws + OFF_DINV);
  _Float16*  adjh  = (_Float16*)(ws + OFF_ADJH);
  _Float16*  BmT   = (_Float16*)(ws + OFF_BMT);
  float*     pre   = (float*)(ws + OFF_PRE);

  // 0) init + weight prep
  zero_hist_k<<<1, 256, 0, stream>>>(hist);
  prep_fcw_k <<<(LPAD * KFC + 255) / 256, 256, 0, stream>>>(fc_w, fcwT);
  prep_gcnw_k<<<(LPAD * LPAD + 255) / 256, 256, 0, stream>>>(gcn_w, gcnwT);
  prep_w2_k  <<<(64 * 512 + 255) / 256, 256, 0, stream>>>(conv2_w, w2T);

  // 1) conv1 (VALU) then conv2 as implicit-GEMM WMMA with async-LDS B panel
  conv1_pool_k<<<(NNODES * 20 * 144 + 255) / 256, 256, 0, stream>>>(x, conv1_w, conv1_b, out1);
  conv2_wmma_k<<<NNODES, 256, 0, stream>>>(out1, w2T, conv2_b, out2h);

  // 2) fc: Hh = relu(out2h @ fc_w + b), f16 [2048,512]   (WMMA, K=800)
  wmma_gemm_k<EPI_RELU_BIAS_F16><<<dim3(NNODES / 64, LPAD / 32), 32, 0, stream>>>(
      out2h, KFC, fcwT, KFC, KFC, nullptr, Hh, LPAD, fc_b, nullptr, LOUT);

  // 3) squared norms
  sq_k<<<(NNODES + 255) / 256, 256, 0, stream>>>(Hh, sq);

  // 4) Gram = Hh @ Hh^T  (WMMA, K=512; reuses conv1 scratch)
  wmma_gemm_k<EPI_F32><<<dim3(NNODES / 64, NNODES / 32), 32, 0, stream>>>(
      Hh, LPAD, Hh, LPAD, LPAD, Gram, nullptr, NNODES, nullptr, nullptr, 0);

  // 5) threshold search: histogram + device scan (replaces host while-loop)
  hist_k<<<(unsigned)(((size_t)NNODES * NNODES + 255) / 256), 256, 0, stream>>>(Gram, sq, hist);
  thr_k<<<1, 32, 0, stream>>>(hist, thr);

  // 6) A_hat f16 + dinv
  adj_k<<<NNODES, 256, 0, stream>>>(Gram, sq, thr, adjh, dinv);

  // 7) BmT[n][j] = dinv[j] * (Hh @ gcn_w)[j][n]  (WMMA + transposed-scaled epilogue)
  wmma_gemm_k<EPI_SCALED_T_F16><<<dim3(NNODES / 64, LPAD / 32), 32, 0, stream>>>(
      Hh, LPAD, gcnwT, LPAD, LPAD, nullptr, BmT, NNODES, nullptr, dinv, 0);

  // 8) pre = dinv[i] * (A_hat @ Bm) + gcn_b  (WMMA, K=2048)
  wmma_gemm_k<EPI_ROWSCALE_BIAS_F32><<<dim3(NNODES / 64, LPAD / 32), 32, 0, stream>>>(
      adjh, NNODES, BmT, NNODES, NNODES, pre, nullptr, LPAD, gcn_b, dinv, LOUT);

  // 9) log-softmax over 500 valid columns
  logsoftmax_k<<<NNODES, 256, 0, stream>>>(pre, out);
}
